// _CausalSelfAttention_11020886082321
// MI455X (gfx1250) — matrix operand order repaired and checked by reading, not compiled, not run on hardware
//
#include <hip/hip_runtime.h>

typedef unsigned short u16;
typedef unsigned int u32;
typedef unsigned long long u64;
typedef __attribute__((ext_vector_type(16))) __bf16 v16bf;
typedef __attribute__((ext_vector_type(8)))  float  v8f;

union FR { uint4 u[2]; v16bf v; };

__device__ __forceinline__ u16 f2bf(float f) {
    u32 u = __float_as_uint(f);
    u32 r = (u + 0x7fffu + ((u >> 16) & 1u)) >> 16;
    return (u16)r;
}

// async copy of one 16B chunk: global -> LDS, tracked by ASYNCcnt
__device__ __forceinline__ void async_b128(const u16* gp, const u16* lp) {
    u32 lds = (u32)(u64)lp;          // low 32 bits of generic shared ptr = LDS offset
    u64 ga  = (u64)gp;
    asm volatile("global_load_async_to_lds_b128 %0, %1, off"
                 :: "v"(lds), "v"(ga) : "memory");
}
__device__ __forceinline__ void wait_async(int n) {
    if (n == 0)      asm volatile("s_wait_asynccnt 0" ::: "memory");
    else if (n == 2) asm volatile("s_wait_asynccnt 2" ::: "memory");
    else             asm volatile("s_wait_asynccnt 4" ::: "memory");
}

// ---------------- conversion kernels ----------------

__global__ void cvt4(const float* __restrict__ s, u16* __restrict__ d, int n4) {
    int i = blockIdx.x * blockDim.x + threadIdx.x;
    if (i < n4) {
        float4 f = ((const float4*)s)[i];
        uint2 o;
        o.x = (u32)f2bf(f.x) | ((u32)f2bf(f.y) << 16);
        o.y = (u32)f2bf(f.z) | ((u32)f2bf(f.w) << 16);
        ((uint2*)d)[i] = o;
    }
}

// w[K][N] (f32) -> wt[N][K] (bf16)
__global__ void cvtT(const float* __restrict__ w, u16* __restrict__ wt,
                     int K, int N, int total) {
    int i = blockIdx.x * blockDim.x + threadIdx.x;
    if (i < total) {
        int n = i % N, k = i / N;
        wt[(size_t)n * K + k] = f2bf(w[i]);
    }
}

// ---------------- bf16 WMMA GEMM: [M,1024] x [1024,N] ----------------
// mode 0: QKV epilogue (scatter q/k/v bf16), mode 1: proj epilogue (f32 + bias)

__global__ __launch_bounds__(256) void gemm_bf16(
    const u16* __restrict__ A,    // [M][1024] bf16 row-major
    const u16* __restrict__ Bt,   // [N][1024] bf16 (B transposed)
    const float* __restrict__ bias,
    int N, int mode,
    u16* __restrict__ qp, u16* __restrict__ kp, u16* __restrict__ vpT,
    float* __restrict__ outp)
{
    __shared__ __align__(16) u16 As[2][128][40];
    __shared__ __align__(16) u16 Bs[2][128][40];
    const int K = 1024;
    const int tid  = threadIdx.x;
    const int wid  = tid >> 5;
    const int lane = tid & 31;
    const int half = lane >> 4;
    const int mr   = lane & 15;
    const int waveM = wid & 3;   // 4 blocks of 32 rows
    const int waveN = wid >> 2;  // 2 blocks of 64 cols
    const int rowBase = blockIdx.y * 128;
    const int colBase = blockIdx.x * 128;

    v8f acc[2][4];
#pragma unroll
    for (int mt = 0; mt < 2; ++mt)
#pragma unroll
        for (int nt = 0; nt < 4; ++nt)
            acc[mt][nt] = (v8f){0.f,0.f,0.f,0.f,0.f,0.f,0.f,0.f};

    // async stage one 128x32 A tile + B tile: 4 chunks per thread per tile
    auto issue_tile = [&](int bf, int kk) {
#pragma unroll
        for (int i = 0; i < 2; ++i) {
            int c = tid + i * 256;               // 512 chunks per matrix tile
            int r = c >> 2, ko = (c & 3) * 8;
            async_b128(&A [(size_t)(rowBase + r) * K + kk + ko], &As[bf][r][ko]);
            async_b128(&Bt[(size_t)(colBase + r) * K + kk + ko], &Bs[bf][r][ko]);
        }
    };

    issue_tile(0, 0);
    const int NK = K / 32;
    for (int kt = 0; kt < NK; ++kt) {
        const int cur = kt & 1;
        if (kt + 1 < NK) {
            issue_tile(cur ^ 1, (kt + 1) * 32);  // overlap copy of next tile
            wait_async(4);                       // current tile complete (in-order)
        } else {
            wait_async(0);
        }
        __syncthreads();

        FR af[2], bf[4];
#pragma unroll
        for (int mt = 0; mt < 2; ++mt) {
            const u16* p = &As[cur][waveM * 32 + mt * 16 + mr][half * 8];
            af[mt].u[0] = *(const uint4*)p;
            af[mt].u[1] = *(const uint4*)(p + 16);
        }
#pragma unroll
        for (int nt = 0; nt < 4; ++nt) {
            const u16* p = &Bs[cur][waveN * 64 + nt * 16 + mr][half * 8];
            bf[nt].u[0] = *(const uint4*)p;
            bf[nt].u[1] = *(const uint4*)(p + 16);
        }
#pragma unroll
        for (int mt = 0; mt < 2; ++mt)
#pragma unroll
            for (int nt = 0; nt < 4; ++nt)
                acc[mt][nt] = __builtin_amdgcn_wmma_f32_16x16x32_bf16(
                    false, af[mt].v, false, bf[nt].v,
                    (short)0, acc[mt][nt], false, false);
        __syncthreads();
    }

    // epilogue
#pragma unroll
    for (int mt = 0; mt < 2; ++mt)
#pragma unroll
        for (int nt = 0; nt < 4; ++nt)
#pragma unroll
            for (int v = 0; v < 8; ++v) {
                int r = rowBase + waveM * 32 + mt * 16 + half * 8 + v;
                int j = colBase + waveN * 64 + nt * 16 + mr;
                float val = acc[mt][nt][v] + bias[j];
                if (mode == 0) {
                    int which = j >> 10, c = j & 1023;
                    int h = c >> 6, d = c & 63;
                    int b = r >> 11, t = r & 2047;
                    size_t bh = (size_t)(b * 16 + h);
                    u16 bv = f2bf(val);
                    if (which == 0)      qp [(bh * 2048 + t) * 64 + d] = bv;
                    else if (which == 1) kp [(bh * 2048 + t) * 64 + d] = bv;
                    else                 vpT[(bh * 64 + d) * 2048 + t] = bv;
                } else {
                    outp[(size_t)r * N + j] = val;
                }
            }
}

// ---------------- fused causal flash attention ----------------
// grid: (T/128, B*H); block: 256 = 8 waves, each wave owns 16 q rows.

__global__ __launch_bounds__(256) void attn(
    const u16* __restrict__ q,   // [b,h,t,d] bf16
    const u16* __restrict__ k,   // [b,h,t,d] bf16
    const u16* __restrict__ vT,  // [b,h,d,t] bf16
    u16* __restrict__ y)         // [b,t,h*d] bf16
{
    __shared__ __align__(16) u16 Ksh[2][32][72];   // 32 keys x 64 d (+pad)
    __shared__ __align__(16) u16 Vsh[2][64][40];   // 64 d x 32 keys (+pad)
    __shared__ __align__(16) u16 Psh[8][16][40];   // per-wave P re-layout buffer
    const int T = 2048;
    const int tid  = threadIdx.x;
    const int wid  = tid >> 5;
    const int lane = tid & 31;
    const int half = lane >> 4;
    const int mr   = lane & 15;
    const int qb = blockIdx.x;
    const int bh = blockIdx.y;
    const int qrow0 = qb * 128 + wid * 16;

    // Q fragments (d 0..31 and d 32..63), loaded once from global
    FR qf[2];
    {
        const u16* p = q + ((size_t)bh * T + qrow0 + mr) * 64;
        qf[0].u[0] = *(const uint4*)(p + half * 8);
        qf[0].u[1] = *(const uint4*)(p + half * 8 + 16);
        qf[1].u[0] = *(const uint4*)(p + 32 + half * 8);
        qf[1].u[1] = *(const uint4*)(p + 32 + half * 8 + 16);
    }

    float mrow[8], lrow[8];
    v8f o[4];
#pragma unroll
    for (int v = 0; v < 8; ++v) { mrow[v] = -1e30f; lrow[v] = 0.f; }
#pragma unroll
    for (int dt = 0; dt < 4; ++dt) o[dt] = (v8f){0.f,0.f,0.f,0.f,0.f,0.f,0.f,0.f};

    // async stage one 32-key K chunk + V^T chunk: 2 chunks per thread
    auto issue_kv = [&](int bf, int k0) {
        int r  = tid >> 3, ko = (tid & 7) * 8;
        async_b128(&k [((size_t)bh * T + k0 + r) * 64 + ko], &Ksh[bf][r][ko]);
        int d  = tid >> 2, jo = (tid & 3) * 8;
        async_b128(&vT[((size_t)bh * 64 + d) * T + k0 + jo], &Vsh[bf][d][jo]);
    };

    const int nchunk = qb * 4 + 4;   // keys up to end of this q block
    issue_kv(0, 0);
    for (int ch = 0; ch < nchunk; ++ch) {
        const int cur = ch & 1;
        if (ch + 1 < nchunk) {
            issue_kv(cur ^ 1, (ch + 1) * 32);    // overlap next K/V copy
            wait_async(2);
        } else {
            wait_async(0);
        }
        __syncthreads();
        const int k0 = ch * 32;

        // S = Q K^T for two 16-key n-tiles, K-dim = 64 (2 wmma each)
        v8f s[2];
#pragma unroll
        for (int nt = 0; nt < 2; ++nt) {
            s[nt] = (v8f){0.f,0.f,0.f,0.f,0.f,0.f,0.f,0.f};
            const u16* p = &Ksh[cur][nt * 16 + mr][0];
            FR kb;
            kb.u[0] = *(const uint4*)(p + half * 8);
            kb.u[1] = *(const uint4*)(p + half * 8 + 16);
            s[nt] = __builtin_amdgcn_wmma_f32_16x16x32_bf16(
                false, qf[0].v, false, kb.v, (short)0, s[nt], false, false);
            kb.u[0] = *(const uint4*)(p + 32 + half * 8);
            kb.u[1] = *(const uint4*)(p + 32 + half * 8 + 16);
            s[nt] = __builtin_amdgcn_wmma_f32_16x16x32_bf16(
                false, qf[1].v, false, kb.v, (short)0, s[nt], false, false);
        }

        // scale + causal mask + online softmax (row = half*8+v, 16 cols across lanes)
#pragma unroll
        for (int v = 0; v < 8; ++v) {
            int trow = qrow0 + half * 8 + v;
            float s0 = s[0][v] * 0.125f; if (k0 + mr      > trow) s0 = -1e30f;
            float s1 = s[1][v] * 0.125f; if (k0 + 16 + mr > trow) s1 = -1e30f;
            float rmax = fmaxf(s0, s1);
            rmax = fmaxf(rmax, __shfl_xor(rmax, 1, 32));
            rmax = fmaxf(rmax, __shfl_xor(rmax, 2, 32));
            rmax = fmaxf(rmax, __shfl_xor(rmax, 4, 32));
            rmax = fmaxf(rmax, __shfl_xor(rmax, 8, 32));
            float mnew = fmaxf(mrow[v], rmax);
            float corr = __expf(mrow[v] - mnew);
            float p0 = __expf(s0 - mnew);
            float p1 = __expf(s1 - mnew);
            float rs = p0 + p1;
            rs += __shfl_xor(rs, 1, 32);
            rs += __shfl_xor(rs, 2, 32);
            rs += __shfl_xor(rs, 4, 32);
            rs += __shfl_xor(rs, 8, 32);
            lrow[v] = lrow[v] * corr + rs;
            mrow[v] = mnew;
#pragma unroll
            for (int dt = 0; dt < 4; ++dt) o[dt][v] *= corr;
            Psh[wid][half * 8 + v][mr]      = f2bf(p0);
            Psh[wid][half * 8 + v][16 + mr] = f2bf(p1);
        }

        // P (A-layout via per-wave LDS) @ V
        FR pf;
        {
            const u16* p = &Psh[wid][mr][half * 8];
            pf.u[0] = *(const uint4*)p;
            pf.u[1] = *(const uint4*)(p + 16);
        }
#pragma unroll
        for (int dt = 0; dt < 4; ++dt) {
            const u16* p = &Vsh[cur][dt * 16 + mr][half * 8];
            FR vf;
            vf.u[0] = *(const uint4*)p;
            vf.u[1] = *(const uint4*)(p + 16);
            o[dt] = __builtin_amdgcn_wmma_f32_16x16x32_bf16(
                false, pf.v, false, vf.v, (short)0, o[dt], false, false);
        }
        __syncthreads();
    }

    // normalize + store y bf16 in [b, t, h*64 + d]
    const int b = bh >> 4, h = bh & 15;
#pragma unroll
    for (int v = 0; v < 8; ++v) {
        float inv = 1.0f / lrow[v];
        int trow = qrow0 + half * 8 + v;
        size_t rowoff = ((size_t)b * T + trow) * 1024 + h * 64;
#pragma unroll
        for (int dt = 0; dt < 4; ++dt)
            y[rowoff + dt * 16 + mr] = f2bf(o[dt][v] * inv);
    }
}

// ---------------- host launch ----------------

extern "C" void kernel_launch(void* const* d_in, const int* in_sizes, int n_in,
                              void* d_out, int out_size, void* d_ws, size_t ws_size,
                              hipStream_t stream) {
    const int B = 4, T = 2048, C = 1024, H = 16;
    const float* x      = (const float*)d_in[0];
    const float* w_qkv  = (const float*)d_in[1];
    const float* b_qkv  = (const float*)d_in[2];
    const float* w_proj = (const float*)d_in[3];
    const float* b_proj = (const float*)d_in[4];
    float* out = (float*)d_out;

    char* ws = (char*)d_ws;
    const size_t BT  = (size_t)B * T;           // 8192
    const size_t BTC = BT * C;                  // 8,388,608
    u16* xb     = (u16*)(ws);                                  // 16 MB
    u16* wqkvT  = (u16*)(ws + 16777216);                       // 6 MB
    u16* wprojT = (u16*)(ws + 16777216 + 6291456);             // 2 MB
    u16* qp     = (u16*)(ws + 25165824);                       // 16 MB
    u16* kp     = (u16*)(ws + 25165824 + 16777216);            // 16 MB
    u16* vpT    = (u16*)(ws + 25165824 + 2 * 16777216);        // 16 MB
    u16* yb     = (u16*)(ws + 25165824 + 3 * 16777216);        // 16 MB

    {   // x -> bf16
        int n4 = (int)(BTC / 4);
        cvt4<<<(n4 + 255) / 256, 256, 0, stream>>>(x, xb, n4);
    }
    {   // w_qkv [1024,3072] -> bf16 [3072,1024]
        int total = C * 3 * C;
        cvtT<<<(total + 255) / 256, 256, 0, stream>>>(w_qkv, wqkvT, C, 3 * C, total);
    }
    {   // w_proj [1024,1024] -> bf16 [1024,1024]^T
        int total = C * C;
        cvtT<<<(total + 255) / 256, 256, 0, stream>>>(w_proj, wprojT, C, C, total);
    }

    // QKV GEMM: [8192,1024] x [1024,3072]
    gemm_bf16<<<dim3(3 * C / 128, (int)(BT / 128)), 256, 0, stream>>>(
        xb, wqkvT, b_qkv, 3 * C, 0, qp, kp, vpT, nullptr);

    // fused causal flash attention
    attn<<<dim3(T / 128, B * H), 256, 0, stream>>>(qp, kp, vpT, yb);

    // proj GEMM: [8192,1024] x [1024,1024] -> f32 out
    gemm_bf16<<<dim3(C / 128, (int)(BT / 128)), 256, 0, stream>>>(
        yb, wprojT, b_proj, C, 1, nullptr, nullptr, nullptr, out);
}
